// Encoder_block_85701777424506
// MI455X (gfx1250) — compile-verified, matrix operand order checked by reading, not run on hardware
//
#include <hip/hip_runtime.h>
#include <math.h>

// ---------------------------------------------------------------------------
// Encoder block (pre-norm attention + MLP) for MI455X / gfx1250, wave32 WMMA
// + Tensor Data Mover tile staging.
// B=8, N=1024, C=1024, H=16, hd=64, HID=4096.
// ---------------------------------------------------------------------------

typedef __bf16 bf16;
typedef __attribute__((ext_vector_type(16))) __bf16    v16bf;
typedef __attribute__((ext_vector_type(8)))  float     v8f;
typedef __attribute__((ext_vector_type(4)))  unsigned  v4u;
typedef __attribute__((ext_vector_type(8)))  int       v8i;
typedef __attribute__((ext_vector_type(4)))  int       v4i;

#define DIM_B   8
#define DIM_N   1024
#define DIM_C   1024
#define DIM_H   16
#define DIM_HD  64
#define DIM_HID 4096
#define MTOK    (DIM_B * DIM_N)      // 8192 rows of tokens

#if defined(__has_builtin)
#if __has_builtin(__builtin_amdgcn_tensor_load_to_lds) && \
    __has_builtin(__builtin_amdgcn_s_wait_tensorcnt)
#define HAVE_TDM 1
#endif
#endif
#ifndef HAVE_TDM
#define HAVE_TDM 0
#endif

// ---------------------------------------------------------------------------
// TDM: 2D bf16 tile load Global -> LDS via Tensor DMA descriptor (ISA 8.3-8.6)
//   group0: [0]=count=1  [1]=lds_addr  [2]=ga[31:0]  [3]=ga[56:32]|type=2
//   group1: [0]=data_size(2B)|pad ctl  dims/strides per bit layout
// Pad feature reproduces the manual "+8 bf16 per row" LDS padding:
//   pad_interval code 3 = every 16 DWORDs (64B = 32 bf16), pad_amount code 3
//   = 4 DWORDs (16B = 8 bf16)  -> row stride 40 bf16.
// ---------------------------------------------------------------------------
#if HAVE_TDM
__device__ __forceinline__ void tdm_load_tile_bf16(
    const void* gptr, unsigned lds_byte_off, unsigned tensor_d0,
    unsigned tensor_d1, unsigned tile_d0, unsigned tile_d1,
    unsigned long long stride0_elems, unsigned pad_interval_code,
    unsigned pad_amount_code, bool pad_en) {
  unsigned long long ga = (unsigned long long)gptr;
  v4u g0;
  g0[0] = 1u;  // count=1, is_restore=0, gather off
  g0[1] = lds_byte_off;
  g0[2] = (unsigned)(ga & 0xFFFFFFFFu);
  g0[3] = (unsigned)((ga >> 32) & 0x01FFFFFFu) | (2u << 30);  // type=2 (image)
  unsigned w0 = (1u << 16);  // workgroup_mask=0, data_size=1 (2 bytes)
  if (pad_en)
    w0 |= (1u << 20) | (pad_interval_code << 22) | (pad_amount_code << 25);
  v8i g1;
  g1[0] = (int)w0;
  g1[1] = (int)((tensor_d0 & 0xFFFFu) << 16);  // atomic_addr=0 | dim0 lo
  g1[2] = (int)((tensor_d0 >> 16) | ((tensor_d1 & 0xFFFFu) << 16));
  g1[3] = (int)((tensor_d1 >> 16) | ((tile_d0 & 0xFFFFu) << 16));
  g1[4] = (int)(tile_d1 & 0xFFFFu);  // tile_dim2 = 0
  g1[5] = (int)(unsigned)(stride0_elems & 0xFFFFFFFFull);
  g1[6] = (int)(unsigned)((stride0_elems >> 32) & 0xFFFFull);  // dim1_stride=0
  g1[7] = 0;
  v4i z4 = {0, 0, 0, 0};
#if __clang_major__ >= 23
  v8i z8 = {0, 0, 0, 0, 0, 0, 0, 0};
  __builtin_amdgcn_tensor_load_to_lds(g0, g1, z4, z4, z8, 0);
#else
  __builtin_amdgcn_tensor_load_to_lds(g0, g1, z4, z4, 0);
#endif
}
// low 32 bits of a generic pointer to __shared__ = LDS byte address
__device__ __forceinline__ unsigned lds_off_u32(const void* p) {
  return (unsigned)(unsigned long long)p;
}
#endif

// ---------------------------------------------------------------------------
// bf16 fragment loader following the CDNA5 16-bit A/B VGPR layout:
//   lane L: row = L%16, half = L/16
//   vector elems (2v, 2v+1) hold K = kpat(v) + 8*half, kpat(v)+1
//   kpat(v) = 2v (v<4) else 8+2v
// Works for A (row = M) and for B when the tile is stored N-major.
// ---------------------------------------------------------------------------
__device__ __forceinline__ v16bf frag_ld(const bf16* __restrict__ base,
                                         int stride, int row, int half) {
  v16bf f;
  const bf16* rp = base + (size_t)row * stride + 8 * half;
#pragma unroll
  for (int v = 0; v < 8; ++v) {
    int k = (v < 4) ? (2 * v) : (8 + 2 * v);
    f[2 * v]     = rp[k];
    f[2 * v + 1] = rp[k + 1];
  }
  return f;
}

__device__ __forceinline__ v8f vzero8() {
  v8f z = {0.f, 0.f, 0.f, 0.f, 0.f, 0.f, 0.f, 0.f};
  return z;
}

// ---------------------------------------------------------------------------
// fp32 -> bf16 elementwise convert (weights / activations)
// ---------------------------------------------------------------------------
__global__ void cvt_f32_bf16(const float* __restrict__ in,
                             bf16* __restrict__ out, long n) {
  long i = (long)blockIdx.x * blockDim.x + threadIdx.x;
  if (i < n) out[i] = (bf16)in[i];
}

// ---------------------------------------------------------------------------
// LayerNorm (row length C=1024) fp32 in -> bf16 out
// ---------------------------------------------------------------------------
__global__ __launch_bounds__(256) void ln_bf16(const float* __restrict__ x,
                                               const float* __restrict__ g,
                                               const float* __restrict__ b,
                                               bf16* __restrict__ out) {
  __shared__ float rs[256], rs2[256];
  const int row = blockIdx.x;
  const float* xr = x + (size_t)row * DIM_C;
  float s = 0.f, s2 = 0.f;
  for (int i = threadIdx.x; i < DIM_C; i += 256) {
    float t = xr[i];
    s += t;
    s2 += t * t;
  }
  rs[threadIdx.x] = s;
  rs2[threadIdx.x] = s2;
  __syncthreads();
  for (int off = 128; off > 0; off >>= 1) {
    if (threadIdx.x < off) {
      rs[threadIdx.x] += rs[threadIdx.x + off];
      rs2[threadIdx.x] += rs2[threadIdx.x + off];
    }
    __syncthreads();
  }
  float mean = rs[0] * (1.0f / DIM_C);
  float var  = rs2[0] * (1.0f / DIM_C) - mean * mean;
  float inv  = rsqrtf(var + 1e-5f);
  for (int i = threadIdx.x; i < DIM_C; i += 256) {
    out[(size_t)row * DIM_C + i] = (bf16)((xr[i] - mean) * inv * g[i] + b[i]);
  }
}

// ---------------------------------------------------------------------------
// Tiled bf16 WMMA GEMM: out[M,N] = A[M,K] @ Bw[K,N] + bias (+resid) (+GELU)
// BM=64 BN=128 BK=32, 256 threads = 8 waves, wave tile 16x64 (4 accumulators).
// A tile staged by the Tensor Data Mover (with HW LDS padding); B tile staged
// by threads (needs the N-major transpose that keeps frag loads ds_load_b128).
// ---------------------------------------------------------------------------
#define BM 64
#define BN 128
#define BK 32

template <bool GELU_ACT, bool RESID, bool OUTF32, bool OUTBF16>
__global__ __launch_bounds__(256) void gemm_bf16_wmma(
    const bf16* __restrict__ A, const bf16* __restrict__ Bw,
    const float* __restrict__ bias, const float* __restrict__ resid,
    float* __restrict__ outF, bf16* __restrict__ outB, int M, int N, int K) {
  __shared__ bf16 As[BM][BK + 8];
  __shared__ bf16 Bs[BN][BK + 8];   // stored N-major (transposed tile)

  const int tid  = threadIdx.x;
  const int lane = tid & 31;
  const int wave = tid >> 5;           // 0..7
  const int wm   = wave >> 1;          // 0..3  (16-row strip)
  const int wn   = wave & 1;           // 0..1  (64-col strip)
  const int half = lane >> 4;
  const int l16  = lane & 15;
  const int mBase = blockIdx.y * BM;
  const int nBase = blockIdx.x * BN;

  v8f acc0 = vzero8(), acc1 = vzero8(), acc2 = vzero8(), acc3 = vzero8();

#if HAVE_TDM
  const unsigned asOff = lds_off_u32(&As[0][0]);
#else
  const int ar = tid >> 2;             // 0..63
  const int ac = (tid & 3) * 8;        // 0,8,16,24
#endif
  const int bk = tid >> 3;             // 0..31
  const int bn = (tid & 7) * 16;       // 0..112

  for (int k0 = 0; k0 < K; k0 += BK) {
#if HAVE_TDM
    // TDM: 64x32 bf16 tile, row stride K, LDS pad 16B per 64B row -> [64][40]
    if (wave == 0) {
      tdm_load_tile_bf16(A + (size_t)mBase * K + k0, asOff, (unsigned)K,
                         (unsigned)M, BK, BM, (unsigned long long)K,
                         /*pad_interval 16dw*/ 3u, /*pad_amount 4dw*/ 3u, true);
    }
#else
    {
      const bf16* src = A + (size_t)(mBase + ar) * K + k0 + ac;
      if (k0 + BK < K) __builtin_prefetch(src + BK, 0, 1);
#pragma unroll
      for (int i = 0; i < 8; ++i) As[ar][ac + i] = src[i];
    }
#endif
    // stage B tile 32x128 transposed -> Bs[n][k]
    {
      const bf16* src = Bw + (size_t)(k0 + bk) * N + nBase + bn;
#pragma unroll
      for (int i = 0; i < 16; ++i) Bs[bn + i][bk] = src[i];
    }
#if HAVE_TDM
    if (wave == 0) __builtin_amdgcn_s_wait_tensorcnt(0);
#endif
    __syncthreads();

    v16bf af = frag_ld(&As[0][0], BK + 8, wm * 16 + l16, half);
    v16bf b0 = frag_ld(&Bs[0][0], BK + 8, wn * 64 + 0 * 16 + l16, half);
    acc0 = __builtin_amdgcn_wmma_f32_16x16x32_bf16(false, af, false, b0,
                                                   (short)0, acc0, false, false);
    v16bf b1 = frag_ld(&Bs[0][0], BK + 8, wn * 64 + 1 * 16 + l16, half);
    acc1 = __builtin_amdgcn_wmma_f32_16x16x32_bf16(false, af, false, b1,
                                                   (short)0, acc1, false, false);
    v16bf b2 = frag_ld(&Bs[0][0], BK + 8, wn * 64 + 2 * 16 + l16, half);
    acc2 = __builtin_amdgcn_wmma_f32_16x16x32_bf16(false, af, false, b2,
                                                   (short)0, acc2, false, false);
    v16bf b3 = frag_ld(&Bs[0][0], BK + 8, wn * 64 + 3 * 16 + l16, half);
    acc3 = __builtin_amdgcn_wmma_f32_16x16x32_bf16(false, af, false, b3,
                                                   (short)0, acc3, false, false);
    __syncthreads();
  }

  v8f accs[4] = {acc0, acc1, acc2, acc3};
#pragma unroll
  for (int c = 0; c < 4; ++c) {
#pragma unroll
    for (int r = 0; r < 8; ++r) {
      int gm = mBase + wm * 16 + r + 8 * half;
      int gn = nBase + wn * 64 + c * 16 + l16;
      float v = accs[c][r] + bias[gn];
      if (RESID) v += resid[(size_t)gm * N + gn];
      if (GELU_ACT) v = 0.5f * v * (1.0f + erff(v * 0.70710678118654752f));
      size_t idx = (size_t)gm * N + gn;
      if (OUTF32)  outF[idx] = v;
      if (OUTBF16) outB[idx] = (bf16)v;
    }
  }
}

// ---------------------------------------------------------------------------
// Attention: one WG per (b, h, 64-row tile). S (64x1024 fp32) kept in LDS
// (CDNA5: 320 KB per WGP). q/k/v are flat (B, N*C) bf16; the reference's
// reshape(B,H,-1,hd) means head h is the contiguous slab h*65536.
// ---------------------------------------------------------------------------
__global__ __launch_bounds__(256) void attn_wmma(const bf16* __restrict__ q,
                                                 const bf16* __restrict__ k,
                                                 const bf16* __restrict__ v,
                                                 bf16* __restrict__ o) {
  extern __shared__ char smem_raw[];
  float* S   = (float*)smem_raw;                                  // 64*1024 f32
  bf16*  Qs  = (bf16*)(smem_raw + 64 * 1024 * 4);                 // 64*64 bf16
  float* red = (float*)(smem_raw + 64 * 1024 * 4 + 64 * 64 * 2);  // 64*4 f32

  const int blk  = blockIdx.x;          // ((b*16 + h)*16 + mt)
  const int mt   = blk & 15;
  const int h    = (blk >> 4) & 15;
  const int b    = blk >> 8;
  const size_t headBase = ((size_t)b * DIM_H + h) * (size_t)(DIM_N * DIM_HD);
  const int m0 = mt * 64;
  const float scale = 0.125f;           // hd^-0.5 = 1/8

  const int tid  = threadIdx.x;
  const int lane = tid & 31;
  const int wave = tid >> 5;
  const int half = lane >> 4;
  const int l16  = lane & 15;

  // stage Q tile (64 rows x 64 d): contiguous 4096-element slab
#if HAVE_TDM
  if (wave == 0) {
    tdm_load_tile_bf16(q + headBase + (size_t)m0 * DIM_HD, lds_off_u32(Qs),
                       4096u, 1u, 4096u, 1u, 4096ull, 0u, 0u, false);
    __builtin_amdgcn_s_wait_tensorcnt(0);
  }
#else
  for (int i = tid; i < 64 * 64; i += 256) {
    int r = i >> 6, c = i & 63;
    Qs[i] = q[headBase + (size_t)(m0 + r) * DIM_HD + c];
  }
#endif
  __syncthreads();

  // ---- Phase 1: S = scale * Q @ K^T  (64 x 1024) ----
  for (int t = wave; t < 4 * 64; t += 8) {
    int rt = t >> 6;      // 0..3
    int ct = t & 63;      // 0..63
    v8f acc = vzero8();
#pragma unroll
    for (int d0 = 0; d0 < DIM_HD; d0 += 32) {
      v16bf af = frag_ld(Qs + d0, DIM_HD, rt * 16 + l16, half);
      // B[d][n] = K[n][d]: row-major K gives contiguous (d,d+1) pairs.
      v16bf bf_ = frag_ld(k + headBase + d0, DIM_HD, ct * 16 + l16, half);
      acc = __builtin_amdgcn_wmma_f32_16x16x32_bf16(false, af, false, bf_,
                                                    (short)0, acc, false, false);
    }
#pragma unroll
    for (int r = 0; r < 8; ++r)
      S[(size_t)(rt * 16 + r + 8 * half) * 1024 + ct * 16 + l16] = acc[r] * scale;
  }
  __syncthreads();

  // ---- Phase 2: row softmax over S (4 threads per row, 256 cols each) ----
  {
    const int row  = tid >> 2;
    const int part = tid & 3;
    float* Srow = S + (size_t)row * 1024 + part * 256;
    float mx = -1e30f;
    for (int i = 0; i < 256; ++i) mx = fmaxf(mx, Srow[i]);
    red[row * 4 + part] = mx;
    __syncthreads();
    mx = fmaxf(fmaxf(red[row * 4 + 0], red[row * 4 + 1]),
               fmaxf(red[row * 4 + 2], red[row * 4 + 3]));
    float sum = 0.f;
    for (int i = 0; i < 256; ++i) {
      float e = __expf(Srow[i] - mx);
      Srow[i] = e;
      sum += e;
    }
    __syncthreads();
    red[row * 4 + part] = sum;
    __syncthreads();
    sum = red[row * 4 + 0] + red[row * 4 + 1] + red[row * 4 + 2] +
          red[row * 4 + 3];
    float inv = 1.0f / sum;
    for (int i = 0; i < 256; ++i) Srow[i] *= inv;
  }
  __syncthreads();

  // ---- Phase 3: O = P @ V  (64 x 64), P converted to bf16 on the fly ----
  for (int t = wave * 2; t < wave * 2 + 2; ++t) {
    int rt = t >> 2;      // 0..3
    int ct = t & 3;       // 0..3
    int dcol = ct * 16 + l16;
    v8f acc = vzero8();
    for (int kk = 0; kk < DIM_N; kk += 32) {
      v16bf af;
      {
        const float* sp =
            S + (size_t)(rt * 16 + l16) * 1024 + kk + 8 * half;
#pragma unroll
        for (int vv = 0; vv < 8; ++vv) {
          int kq = (vv < 4) ? (2 * vv) : (8 + 2 * vv);
          af[2 * vv]     = (bf16)sp[kq];
          af[2 * vv + 1] = (bf16)sp[kq + 1];
        }
      }
      v16bf bfr;
      {
        const bf16* vb =
            v + headBase + dcol + (size_t)(kk + 8 * half) * DIM_HD;
#pragma unroll
        for (int vv = 0; vv < 8; ++vv) {
          int kq = (vv < 4) ? (2 * vv) : (8 + 2 * vv);
          bfr[2 * vv]     = vb[(size_t)kq * DIM_HD];
          bfr[2 * vv + 1] = vb[(size_t)(kq + 1) * DIM_HD];
        }
      }
      acc = __builtin_amdgcn_wmma_f32_16x16x32_bf16(false, af, false, bfr,
                                                    (short)0, acc, false, false);
    }
    // o[b][m][h*64 + d]  (swapaxes(1,2).reshape)
#pragma unroll
    for (int r = 0; r < 8; ++r) {
      int m = m0 + rt * 16 + r + 8 * half;
      o[((size_t)b * DIM_N + m) * DIM_C + h * DIM_HD + dcol] = (bf16)acc[r];
    }
  }
}

// ---------------------------------------------------------------------------
// Host-side orchestration
// ---------------------------------------------------------------------------
extern "C" void kernel_launch(void* const* d_in, const int* in_sizes, int n_in,
                              void* d_out, int out_size, void* d_ws,
                              size_t ws_size, hipStream_t stream) {
  (void)in_sizes; (void)n_in; (void)out_size; (void)ws_size;

  const float* x  = (const float*)d_in[0];
  const float* g1 = (const float*)d_in[1];
  const float* b1 = (const float*)d_in[2];
  const float* Wq = (const float*)d_in[3];
  const float* bq = (const float*)d_in[4];
  const float* Wk = (const float*)d_in[5];
  const float* bk = (const float*)d_in[6];
  const float* Wv = (const float*)d_in[7];
  const float* bv = (const float*)d_in[8];
  const float* Wp = (const float*)d_in[9];
  const float* bp = (const float*)d_in[10];
  const float* g2 = (const float*)d_in[11];
  const float* b2 = (const float*)d_in[12];
  const float* W1 = (const float*)d_in[13];
  const float* c1 = (const float*)d_in[14];
  const float* W2 = (const float*)d_in[15];
  const float* c2 = (const float*)d_in[16];

  // workspace partition
  char* p = (char*)d_ws;
  auto alloc = [&](size_t bytes) {
    char* r = p;
    p += (bytes + 255) & ~(size_t)255;
    return r;
  };
  const size_t CC = (size_t)DIM_C * DIM_C;           // 1M
  const size_t CH = (size_t)DIM_C * DIM_HID;         // 4M
  const size_t TC = (size_t)MTOK * DIM_C;            // 8M
  const size_t TH = (size_t)MTOK * DIM_HID;          // 32M

  bf16*  wq_b  = (bf16*)alloc(CC * 2);
  bf16*  wk_b  = (bf16*)alloc(CC * 2);
  bf16*  wv_b  = (bf16*)alloc(CC * 2);
  bf16*  wp_b  = (bf16*)alloc(CC * 2);
  bf16*  w1_b  = (bf16*)alloc(CH * 2);
  bf16*  w2_b  = (bf16*)alloc(CH * 2);
  bf16*  h_b   = (bf16*)alloc(TC * 2);    // ln1(x)
  bf16*  q_b   = (bf16*)alloc(TC * 2);
  bf16*  k_b   = (bf16*)alloc(TC * 2);
  bf16*  v_b   = (bf16*)alloc(TC * 2);
  bf16*  o_b   = (bf16*)alloc(TC * 2);    // attention output
  float* xres  = (float*)alloc(TC * 4);   // x + proj
  bf16*  h2_b  = (bf16*)alloc(TC * 2);    // ln2(xres)
  bf16*  t1_b  = (bf16*)alloc(TH * 2);    // gelu(h2@W1+c1)

  // weight converts fp32 -> bf16
  cvt_f32_bf16<<<(int)((CC + 255) / 256), 256, 0, stream>>>(Wq, wq_b, (long)CC);
  cvt_f32_bf16<<<(int)((CC + 255) / 256), 256, 0, stream>>>(Wk, wk_b, (long)CC);
  cvt_f32_bf16<<<(int)((CC + 255) / 256), 256, 0, stream>>>(Wv, wv_b, (long)CC);
  cvt_f32_bf16<<<(int)((CC + 255) / 256), 256, 0, stream>>>(Wp, wp_b, (long)CC);
  cvt_f32_bf16<<<(int)((CH + 255) / 256), 256, 0, stream>>>(W1, w1_b, (long)CH);
  cvt_f32_bf16<<<(int)((CH + 255) / 256), 256, 0, stream>>>(W2, w2_b, (long)CH);

  // LN1
  ln_bf16<<<MTOK, 256, 0, stream>>>(x, g1, b1, h_b);

  // QKV projections: (8192 x 1024) = h @ W + b
  dim3 gC(DIM_C / BN, MTOK / BM);   // (8, 128)
  gemm_bf16_wmma<false, false, false, true><<<gC, 256, 0, stream>>>(
      h_b, wq_b, bq, nullptr, nullptr, q_b, MTOK, DIM_C, DIM_C);
  gemm_bf16_wmma<false, false, false, true><<<gC, 256, 0, stream>>>(
      h_b, wk_b, bk, nullptr, nullptr, k_b, MTOK, DIM_C, DIM_C);
  gemm_bf16_wmma<false, false, false, true><<<gC, 256, 0, stream>>>(
      h_b, wv_b, bv, nullptr, nullptr, v_b, MTOK, DIM_C, DIM_C);

  // attention: B*H*16 workgroups, 271,360 B dynamic LDS (fits 320 KB/WGP)
  const size_t attn_lds = 64 * 1024 * 4 + 64 * 64 * 2 + 64 * 4 * 4;
  attn_wmma<<<DIM_B * DIM_H * 16, 256, attn_lds, stream>>>(q_b, k_b, v_b, o_b);

  // projection + residual: xres = x + o@Wp + bp  (fp32)
  gemm_bf16_wmma<false, true, true, false><<<gC, 256, 0, stream>>>(
      o_b, wp_b, bp, x, xres, nullptr, MTOK, DIM_C, DIM_C);

  // LN2
  ln_bf16<<<MTOK, 256, 0, stream>>>(xres, g2, b2, h2_b);

  // MLP up + exact GELU: t1 = gelu(h2 @ W1 + c1)  (bf16, 8192 x 4096)
  dim3 gH(DIM_HID / BN, MTOK / BM);  // (32, 128)
  gemm_bf16_wmma<true, false, false, true><<<gH, 256, 0, stream>>>(
      h2_b, w1_b, c1, nullptr, nullptr, t1_b, MTOK, DIM_HID, DIM_C);

  // MLP down + residual: out = xres + t1 @ W2 + c2  (fp32, final output)
  gemm_bf16_wmma<false, true, true, false><<<gC, 256, 0, stream>>>(
      t1_b, w2_b, c2, xres, (float*)d_out, nullptr, MTOK, DIM_C, DIM_HID);
}
